// UniPhyLayer_54443005444289
// MI455X (gfx1250) — matrix operand order repaired and checked by reading, not compiled, not run on hardware
//
#include <hip/hip_runtime.h>
#include <hip/hip_bf16.h>

typedef __attribute__((ext_vector_type(2)))  float        v2f;
typedef __attribute__((ext_vector_type(8)))  float        v8f;
typedef __attribute__((ext_vector_type(16))) _Float16     v16h;
typedef __attribute__((ext_vector_type(4)))  unsigned int u32x4;
typedef __attribute__((ext_vector_type(4)))  int          i32x4;
typedef __attribute__((ext_vector_type(8)))  int          i32x8;

// ---------------------------------------------------------------------------
// WMMA wrappers (CDNA5 / gfx1250, wave32)
// ---------------------------------------------------------------------------
__device__ __forceinline__ v8f wmma_f32_k4(v2f a, v2f b, v8f c) {
  // D = A(16x4 f32) * B(4x16 f32) + C(16x16 f32)
  return __builtin_amdgcn_wmma_f32_16x16x4_f32(
      false, a, false, b, (short)0, c, false, false);
}

__device__ __forceinline__ v8f wmma_f16_k32(v16h a, v16h b, v8f c) {
  // D = A(16x32 f16) * B(32x16 f16) + C(16x16 f32)
  return __builtin_amdgcn_wmma_f32_16x16x32_f16(
      false, a, false, b, (short)0, c, false, false);
}

// ---------------------------------------------------------------------------
// Tensor Data Mover: DMA a rows x cols fp32 tile (row stride in elements)
// from global memory into LDS.  D# per CDNA5 ISA ch.8 (group0/group1).
// Issued by one wave; caller must s_wait_tensorcnt + barrier.
// ---------------------------------------------------------------------------
#if __has_builtin(__builtin_amdgcn_tensor_load_to_lds)
#define UPL_HAVE_TDM 1
__device__ __forceinline__ void tdm_load_2d_f32(const float* gptr,
                                                void* lds_ptr, int rows,
                                                int cols, int stride) {
  const unsigned long long ga = (unsigned long long)(uintptr_t)gptr;
  // low 32 bits of a flat LDS pointer are the LDS byte offset
  const unsigned int lds = (unsigned int)(uintptr_t)lds_ptr;

  u32x4 g0 = {};
  g0[0] = 1u;                                   // count=1 (valid descriptor)
  g0[1] = lds;                                  // lds_addr
  g0[2] = (unsigned int)(ga & 0xFFFFFFFFull);   // global_addr[31:0]
  g0[3] = (unsigned int)((ga >> 32) & 0x01FFFFFFull) | (2u << 30);  // type=2

  i32x8 g1 = {};
  g1[0] = (2 << 16);                            // data_size=2 -> 4 bytes
  g1[1] = (cols & 0xFFFF) << 16;                // tensor_dim0[15:0]
  g1[2] = ((cols >> 16) & 0xFFFF) |             // tensor_dim0[31:16]
          ((rows & 0xFFFF) << 16);              // tensor_dim1[15:0]
  g1[3] = ((rows >> 16) & 0xFFFF) |             // tensor_dim1[31:16]
          ((cols & 0xFFFF) << 16);              // tile_dim0
  g1[4] = (rows & 0xFFFF);                      // tile_dim1 (tile_dim2 = 0)
  g1[5] = stride;                               // tensor_dim0_stride[31:0]
  g1[6] = 0;                                    // stride hi / dim1_stride
  g1[7] = 0;

  i32x4 g2 = {};                                // 2D tensor: groups 2/3 unused
  i32x4 g3 = {};
#if __clang_major__ >= 23
  i32x8 g4 = {};
  __builtin_amdgcn_tensor_load_to_lds(g0, g1, g2, g3, g4, 0);
#else
  __builtin_amdgcn_tensor_load_to_lds(g0, g1, g2, g3, 0);
#endif
}
#else
#define UPL_HAVE_TDM 0
#endif

// ---------------------------------------------------------------------------
// DFT matrix init: F[k][n] = exp(-2*pi*i*k*n/64)/8   (ortho norm, symmetric)
// ---------------------------------------------------------------------------
__global__ __launch_bounds__(64) void init_dft_kernel(float* __restrict__ Fre,
                                                      float* __restrict__ Fim) {
  int i = blockIdx.x * 64 + threadIdx.x;   // 0..4095
  int k = i >> 6, n = i & 63;
  float ang = -2.0f * 3.14159265358979323846f * (float)(k * n) / 64.0f;
  Fre[i] = cosf(ang) * 0.125f;
  Fim[i] = sinf(ang) * 0.125f;
}

// ---------------------------------------------------------------------------
// Weight repack to f16, layout [tap][kblk][cout][k32] so B-fragment loads are
// contiguous 32B per lane; plus a negated copy of wb for the -conv(b,wb) term
// ---------------------------------------------------------------------------
__global__ __launch_bounds__(256) void weights_to_f16_kernel(
    const float* __restrict__ wa, const float* __restrict__ wb,
    _Float16* __restrict__ oA, _Float16* __restrict__ oB,
    _Float16* __restrict__ oBn) {
  int idx = blockIdx.x * 256 + threadIdx.x;      // 9*64*64 = 36864
  if (idx >= 9 * 64 * 64) return;
  int co = idx & 63, ci = (idx >> 6) & 63, tap = idx >> 12;
  int ky = tap / 3, kx = tap % 3;
  int src = ((co * 64 + ci) * 3 + ky) * 3 + kx;  // OIHW
  int dst = (((tap * 2 + (ci >> 5)) * 64) + co) * 32 + (ci & 31);
  oA[dst]  = (_Float16)wa[src];
  float vb = wb[src];
  oB[dst]  = (_Float16)vb;
  oBn[dst] = (_Float16)(-vb);
}

// ---------------------------------------------------------------------------
// f32 -> f16 field conversion
// ---------------------------------------------------------------------------
__global__ __launch_bounds__(256) void f32_to_f16_kernel(
    const float* __restrict__ in, _Float16* __restrict__ out, int n) {
  int idx = blockIdx.x * 256 + threadIdx.x;
  if (idx < n) out[idx] = (_Float16)in[idx];
}

// ---------------------------------------------------------------------------
// Per-(b,t) channel means + params = stats @ gen_w + gen_b  (128 -> 256)
// ---------------------------------------------------------------------------
__global__ __launch_bounds__(256) void stats_params_kernel(
    const float* __restrict__ x, const float* __restrict__ gen_w,
    const float* __restrict__ gen_b, float* __restrict__ params) {
  __shared__ float stats[128];
  __shared__ float red[256];
  const int bt = blockIdx.x;     // 0..31
  const int tid = threadIdx.x;
  for (int c = 0; c < 128; ++c) {
    float s = 0.f;
    const float* px = x + ((size_t)bt * 128 + c) * 4096;
    for (int i = tid; i < 4096; i += 256) s += px[i];
    red[tid] = s;
    __syncthreads();
    for (int o = 128; o; o >>= 1) {
      if (tid < o) red[tid] += red[tid + o];
      __syncthreads();
    }
    if (tid == 0) stats[c] = red[0] * (1.0f / 4096.0f);
    __syncthreads();
  }
  float acc = gen_b[tid];
  for (int c = 0; c < 128; ++c) acc += stats[c] * gen_w[c * 256 + tid];
  params[bt * 256 + tid] = acc;
}

// ---------------------------------------------------------------------------
// 2D DFT via two fp32-WMMA complex GEMM passes:  Y = (F . X) . F
// (F symmetric so F^T == F; inverse = conjugate, INV template).
// One workgroup (8 waves) per 64x64 image; 16 tiles, 2 per wave.
// Input tile staged into LDS by the Tensor Data Mover when available.
// ---------------------------------------------------------------------------
template <bool CIN, bool COUT, bool INV>
__global__ __launch_bounds__(256) void dft2d_kernel(
    const float* __restrict__ in_re, const float* __restrict__ in_im,
    float* __restrict__ out_re, float* __restrict__ out_im,
    const float* __restrict__ Fre, const float* __restrict__ Fim) {
  __shared__ float Lre[64][64];
  __shared__ float Lim[64][64];
  constexpr float csign = INV ? -1.0f : 1.0f;
  const size_t base = (size_t)blockIdx.x * 4096;
  const int tid = threadIdx.x;

#if UPL_HAVE_TDM
  if (tid < 32) {                       // one wave issues the DMA(s)
    tdm_load_2d_f32(in_re + base, &Lre[0][0], 64, 64, 64);
    if (CIN) tdm_load_2d_f32(in_im + base, &Lim[0][0], 64, 64, 64);
    __builtin_amdgcn_s_wait_tensorcnt(0);
  }
#else
  for (int i = tid; i < 4096; i += 256) {
    Lre[i >> 6][i & 63] = in_re[base + i];
    if (CIN) Lim[i >> 6][i & 63] = in_im[base + i];
  }
#endif
  __syncthreads();

  const int wave = tid >> 5, lane = tid & 31;
  const int rr = lane & 15;             // row (A) / col (B,C,D) within tile
  const int kb = (lane >> 4) << 1;      // fp32 A/B frag K base: 0 or 2
  const int mh = (lane >> 4) << 3;      // C/D frag M offset: 0 or 8

  int mt[2], nt[2];
  v8f yre[2], yim[2];
  const v8f vzero = {};
#pragma unroll
  for (int i = 0; i < 2; ++i) {
    int t = wave * 2 + i;
    mt[i] = t >> 2; nt[i] = t & 3;
    yre[i] = vzero; yim[i] = vzero;
  }

  // Pass 1: T = F . X   (A = F from global, B = X from LDS)
  for (int k4 = 0; k4 < 64; k4 += 4) {
#pragma unroll
    for (int i = 0; i < 2; ++i) {
      const int mrow = mt[i] * 16 + rr;
      v2f afr, afi, nfi;
      afr[0] = Fre[mrow * 64 + k4 + kb];
      afr[1] = Fre[mrow * 64 + k4 + kb + 1];
      afi[0] = csign * Fim[mrow * 64 + k4 + kb];
      afi[1] = csign * Fim[mrow * 64 + k4 + kb + 1];
      nfi[0] = -afi[0]; nfi[1] = -afi[1];
      const int col = nt[i] * 16 + rr;
      v2f bxr;
      bxr[0] = Lre[k4 + kb][col];     bxr[1] = Lre[k4 + kb + 1][col];
      yre[i] = wmma_f32_k4(afr, bxr, yre[i]);
      yim[i] = wmma_f32_k4(afi, bxr, yim[i]);
      if (CIN) {                      // complex input: two more terms
        v2f bxi;
        bxi[0] = Lim[k4 + kb][col];   bxi[1] = Lim[k4 + kb + 1][col];
        yre[i] = wmma_f32_k4(nfi, bxi, yre[i]);
        yim[i] = wmma_f32_k4(afr, bxi, yim[i]);
      }
    }
  }
  __syncthreads();                    // everyone done reading X
#pragma unroll
  for (int i = 0; i < 2; ++i) {
#pragma unroll
    for (int r = 0; r < 8; ++r) {
      const int M = mt[i] * 16 + r + mh;
      const int N = nt[i] * 16 + rr;
      Lre[M][N] = yre[i][r];
      Lim[M][N] = yim[i][r];
    }
  }
  __syncthreads();

  // Pass 2: Y = T . F   (A = T from LDS, B = F from global)
  v8f zre[2], zim[2];
#pragma unroll
  for (int i = 0; i < 2; ++i) { zre[i] = vzero; zim[i] = vzero; }
  for (int k4 = 0; k4 < 64; k4 += 4) {
#pragma unroll
    for (int i = 0; i < 2; ++i) {
      const int mrow = mt[i] * 16 + rr;
      v2f atr, ati;
      atr[0] = Lre[mrow][k4 + kb];    atr[1] = Lre[mrow][k4 + kb + 1];
      ati[0] = Lim[mrow][k4 + kb];    ati[1] = Lim[mrow][k4 + kb + 1];
      const int col = nt[i] * 16 + rr;
      v2f bfr, bfi, nbfi;
      bfr[0] = Fre[(k4 + kb) * 64 + col];
      bfr[1] = Fre[(k4 + kb + 1) * 64 + col];
      bfi[0] = csign * Fim[(k4 + kb) * 64 + col];
      bfi[1] = csign * Fim[(k4 + kb + 1) * 64 + col];
      nbfi[0] = -bfi[0]; nbfi[1] = -bfi[1];
      zre[i] = wmma_f32_k4(atr, bfr, zre[i]);
      zre[i] = wmma_f32_k4(ati, nbfi, zre[i]);
      if (COUT) {
        zim[i] = wmma_f32_k4(atr, bfi, zim[i]);
        zim[i] = wmma_f32_k4(ati, bfr, zim[i]);
      }
    }
  }
#pragma unroll
  for (int i = 0; i < 2; ++i) {
#pragma unroll
    for (int r = 0; r < 8; ++r) {
      const int M = mt[i] * 16 + r + mh;
      const int N = nt[i] * 16 + rr;
      out_re[base + M * 64 + N] = zre[i][r];
      if (COUT) out_im[base + M * 64 + N] = zim[i][r];
    }
  }
}

// ---------------------------------------------------------------------------
// Diagonal SSM scan over T=16 in Fourier space (in-place on spectrum)
// ---------------------------------------------------------------------------
__global__ __launch_bounds__(256) void scan_kernel(
    float* __restrict__ sre, float* __restrict__ sim,
    const float* __restrict__ params, const float* __restrict__ dts) {
  const int idx = blockIdx.x * 256 + threadIdx.x;  // 2*128*4096 threads
  const int pos = idx & 4095;
  const int c   = (idx >> 12) & 127;
  const int b   = idx >> 19;
  float hr = 0.f, hi = 0.f;
  for (int t = 0; t < 16; ++t) {
    const int btc = b * 16 + t;
    const float nu = params[btc * 256 + c];
    const float om = params[btc * 256 + 128 + c];
    const float dt = dts[btc];
    const float lr = -fabsf(nu), li = om;
    const float s  = expf(lr * dt);
    const float Ar = s * cosf(li * dt);
    const float Ai = s * sinf(li * dt);
    const float mag2 = lr * lr + li * li;
    float Xtr, Xti;
    if (mag2 < 1e-12f) {               // |lam| < EPS_LAM  ->  X_term = dt
      Xtr = dt; Xti = 0.f;
    } else {                           // (exp(lam*dt)-1)/lam
      const float inv = 1.0f / mag2;
      const float nr = Ar - 1.0f, ni = Ai;
      Xtr = (nr * lr + ni * li) * inv;
      Xti = (ni * lr - nr * li) * inv;
    }
    const size_t off = ((size_t)btc * 128 + c) * 4096 + pos;
    const float xr = sre[off], xi = sim[off];
    const float Xr = xr * Xtr - xi * Xti;
    const float Xi = xr * Xti + xi * Xtr;
    const float nhr = Ar * hr - Ai * hi + Xr;
    const float nhi = Ar * hi + Ai * hr + Xi;
    hr = nhr; hi = nhi;
    sre[off] = hr; sim[off] = hi;      // in-place: becomes h_fft
  }
}

// ---------------------------------------------------------------------------
// Clifford conv as implicit GEMM with f16 WMMA.
// Block = 8 waves; wave -> (oa/ob half, 16-out-channel block).
// u = hf + clifford_conv(hf) + bias
// ---------------------------------------------------------------------------
__global__ __launch_bounds__(256) void clifford_kernel(
    const _Float16* __restrict__ h16, const float* __restrict__ hf,
    const _Float16* __restrict__ wA, const _Float16* __restrict__ wB,
    const _Float16* __restrict__ wBn,
    const float* __restrict__ cba, const float* __restrict__ cbb,
    float* __restrict__ u) {
  __shared__ _Float16 tile[3][18][128];          // rows x (16+halo) x cin
  const int seg = blockIdx.x;                    // 0..3  (16-wide W segment)
  const int row = blockIdx.y;                    // 0..63
  const int bt  = blockIdx.z;                    // 0..31
  const int tid = threadIdx.x;

  __builtin_prefetch(wA, 0, 3);                  // global_prefetch_b8

  for (int i = tid; i < 3 * 18 * 128; i += 256) {
    const int c  = i & 127;
    const int r2 = i >> 7;
    const int xx = r2 % 18;
    const int dy = r2 / 18;
    const int gy = row + dy - 1;
    const int gx = seg * 16 + xx - 1;
    _Float16 v = (_Float16)0.0f;
    if ((unsigned)gy < 64u && (unsigned)gx < 64u)
      v = h16[(((size_t)bt * 128 + c) * 64 + gy) * 64 + gx];
    tile[dy][xx][c] = v;
  }
  __syncthreads();

  const int wave = tid >> 5, lane = tid & 31;
  const int oh  = wave >> 2;                     // 0 = oa, 1 = ob
  const int cb  = wave & 3;                      // 16-out-channel block
  const int m   = lane & 15;                     // A row / B,C,D col
  const int khi = lane >> 4;

  v8f acc = {};
  for (int tap = 0; tap < 9; ++tap) {
    const int dy = tap / 3, dx = tap % 3;
    for (int sh = 0; sh < 2; ++sh) {             // source half (a, b)
      // oa = conv(a,wa) - conv(b,wb); ob = conv(a,wb) + conv(b,wa)
      const _Float16* wsel =
          (oh == 0) ? (sh == 0 ? wA : wBn) : (sh == 0 ? wB : wA);
      for (int kb2 = 0; kb2 < 2; ++kb2) {        // two K-blocks of 32 cin
        v16h af, bf;
#pragma unroll
        for (int h = 0; h < 16; ++h) {
          const int kkA = khi * 16 + h;   // A 16x32 f16
          af[h] = tile[dy][m + dx][sh * 64 + kb2 * 32 + kkA];
          // B layout [tap][kblk][cout][k32]: contiguous 16 halves per lane
          bf[h] = wsel[(((size_t)tap * 2 + kb2) * 64 + cb * 16 + m) * 32 +
                       khi * 16 + h];
        }
        acc = wmma_f16_k32(af, bf, acc);
      }
    }
  }

  const int co = oh * 64 + cb * 16 + m;          // output channel (N = m)
  const float bv = (oh == 0) ? cba[cb * 16 + m] : cbb[cb * 16 + m];
#pragma unroll
  for (int r = 0; r < 8; ++r) {
    const int px = r + (khi << 3);               // pixel (M) within segment
    const size_t o =
        (((size_t)bt * 128 + co) * 64 + row) * 64 + seg * 16 + px;
    u[o] = acc[r] + hf[o] + bv;                  // residual + bias
  }
}

// ---------------------------------------------------------------------------
// Helmholtz projection, pointwise in full-complex Fourier space (in-place)
// ---------------------------------------------------------------------------
__global__ __launch_bounds__(256) void helm_kernel(float* __restrict__ re,
                                                   float* __restrict__ im) {
  const int idx = blockIdx.x * 256 + threadIdx.x;  // 32*64*4096 threads
  const int pos = idx & 4095;
  const int c   = (idx >> 12) & 63;
  const int bt  = idx >> 18;
  const int wq = pos & 63, hq = pos >> 6;
  const float kx = (float)(wq < 32 ? wq : wq - 64) * (1.0f / 64.0f);
  const float ky = (float)(hq < 32 ? hq : hq - 64) * (1.0f / 64.0f);
  const float k2 = kx * kx + ky * ky;
  const size_t ox = ((size_t)bt * 128 + c) * 4096 + pos;
  const size_t oy = ((size_t)bt * 128 + 64 + c) * 4096 + pos;
  const float vxr = re[ox], vxi = im[ox];
  const float vyr = re[oy], vyi = im[oy];
  float pr = 0.f, pi = 0.f;
  if (k2 > 0.f) {
    const float inv = 1.0f / k2;
    pr = (kx * vxr + ky * vyr) * inv;
    pi = (kx * vxi + ky * vyi) * inv;
  }
  re[ox] = vxr - kx * pr;  im[ox] = vxi - kx * pi;
  re[oy] = vyr - ky * pr;  im[oy] = vyi - ky * pi;
}

// ---------------------------------------------------------------------------
// GroupNorm (4 groups of 32 channels), one block per (bt, group)
// ---------------------------------------------------------------------------
__global__ __launch_bounds__(256) void gn_kernel(
    const float* __restrict__ u, const float* __restrict__ gw,
    const float* __restrict__ gb, float* __restrict__ out) {
  __shared__ float s1[256], s2[256];
  const int bt = blockIdx.x, g = blockIdx.y;
  const int tid = threadIdx.x;
  const size_t base = ((size_t)bt * 128 + g * 32) * 4096;
  const int n = 32 * 4096;
  float sum = 0.f, ss = 0.f;
  for (int i = tid; i < n; i += 256) {
    const float v = u[base + i];
    sum += v; ss += v * v;
  }
  s1[tid] = sum; s2[tid] = ss;
  __syncthreads();
  for (int o = 128; o; o >>= 1) {
    if (tid < o) { s1[tid] += s1[tid + o]; s2[tid] += s2[tid + o]; }
    __syncthreads();
  }
  const float mean = s1[0] / (float)n;
  const float var  = s2[0] / (float)n - mean * mean;
  const float rinv = rsqrtf(var + 1e-5f);
  for (int i = tid; i < n; i += 256) {
    const int c = g * 32 + (i >> 12);
    out[base + i] = (u[base + i] - mean) * rinv * gw[c] + gb[c];
  }
}

// ---------------------------------------------------------------------------
// Host launcher
// ---------------------------------------------------------------------------
extern "C" void kernel_launch(void* const* d_in, const int* in_sizes, int n_in,
                              void* d_out, int out_size, void* d_ws,
                              size_t ws_size, hipStream_t stream) {
  const float* x_seq  = (const float*)d_in[0];
  const float* dt_seq = (const float*)d_in[1];
  const float* gen_w  = (const float*)d_in[2];
  const float* gen_b  = (const float*)d_in[3];
  const float* cw_a   = (const float*)d_in[4];
  const float* cw_b   = (const float*)d_in[5];
  const float* cb_a   = (const float*)d_in[6];
  const float* cb_b   = (const float*)d_in[7];
  const float* gn_w   = (const float*)d_in[8];
  const float* gn_b   = (const float*)d_in[9];
  float* out = (float*)d_out;

  char* p = (char*)d_ws;
  auto alloc = [&](size_t bytes) -> char* {
    char* r = p;
    p += (bytes + 255) & ~(size_t)255;
    return r;
  };
  const size_t NELEM = (size_t)32 * 128 * 4096;  // 16,777,216
  float*    Fre    = (float*)alloc(4096 * sizeof(float));
  float*    Fim    = (float*)alloc(4096 * sizeof(float));
  float*    params = (float*)alloc(32 * 256 * sizeof(float));
  _Float16* wA     = (_Float16*)alloc(36864 * sizeof(_Float16));
  _Float16* wB     = (_Float16*)alloc(36864 * sizeof(_Float16));
  _Float16* wBn    = (_Float16*)alloc(36864 * sizeof(_Float16));
  float*    sre    = (float*)alloc(NELEM * sizeof(float));
  float*    sim    = (float*)alloc(NELEM * sizeof(float));
  float*    hf     = (float*)alloc(NELEM * sizeof(float));
  _Float16* h16    = (_Float16*)alloc(NELEM * sizeof(_Float16));
  float*    u      = (float*)alloc(NELEM * sizeof(float));

  init_dft_kernel<<<64, 64, 0, stream>>>(Fre, Fim);
  weights_to_f16_kernel<<<(36864 + 255) / 256, 256, 0, stream>>>(
      cw_a, cw_b, wA, wB, wBn);
  stats_params_kernel<<<32, 256, 0, stream>>>(x_seq, gen_w, gen_b, params);

  // forward fft2 of x_seq (real -> complex)
  dft2d_kernel<false, true, false><<<4096, 256, 0, stream>>>(
      x_seq, nullptr, sre, sim, Fre, Fim);
  // modal scan over T (in place, spectrum -> h_fft)
  scan_kernel<<<(2 * 128 * 4096) / 256, 256, 0, stream>>>(sre, sim, params,
                                                          dt_seq);
  // inverse fft2 (complex -> real)
  dft2d_kernel<true, false, true><<<4096, 256, 0, stream>>>(
      sre, sim, hf, nullptr, Fre, Fim);
  // f16 copy of h for the WMMA conv
  f32_to_f16_kernel<<<(int)((NELEM + 255) / 256), 256, 0, stream>>>(
      hf, h16, (int)NELEM);
  // Clifford conv + residual + bias
  dim3 cg(4, 64, 32);
  clifford_kernel<<<cg, 256, 0, stream>>>(h16, hf, wA, wB, wBn, cb_a, cb_b, u);
  // Helmholtz: forward fft2 of u, projection, inverse fft2
  dft2d_kernel<false, true, false><<<4096, 256, 0, stream>>>(
      u, nullptr, sre, sim, Fre, Fim);
  helm_kernel<<<(32 * 64 * 4096) / 256, 256, 0, stream>>>(sre, sim);
  dft2d_kernel<true, false, true><<<4096, 256, 0, stream>>>(
      sre, sim, u, nullptr, Fre, Fim);
  // GroupNorm -> output
  dim3 gg(32, 4);
  gn_kernel<<<gg, 256, 0, stream>>>(u, gn_w, gn_b, out);
}
